// MultiHeadSelfAttention_850403524624
// MI455X (gfx1250) — matrix operand order repaired and checked by reading, not compiled, not run on hardware
//
#include <hip/hip_runtime.h>
#include <hip/hip_bf16.h>

typedef __attribute__((ext_vector_type(16))) __bf16 v16bf;
typedef __attribute__((ext_vector_type(8)))  __bf16 v8bf;
typedef __attribute__((ext_vector_type(8)))  float  v8f;
typedef __attribute__((ext_vector_type(4)))  int    v4i;

#define N_TOK   4096
#define D_MODEL 1024
#define N_HEADS 8
#define D_HEAD  128

// ---------------------------------------------------------------------------
// CDNA5 async global->LDS copy (ASYNCcnt path), with safe fallback.
// Builtin signature (leaked by clang diagnostic): (V4i addrspace(1)*,
// V4i addrspace(3)*, imm offset, imm cpol).
// ---------------------------------------------------------------------------
#if defined(__AMDGCN__) && __has_builtin(__builtin_amdgcn_global_load_async_to_lds_b128)
#define HAVE_ASYNC_LDS 1
typedef v4i __attribute__((address_space(1)))* gptr_v4i;
typedef v4i __attribute__((address_space(3)))* lptr_v4i;
#endif

__device__ __forceinline__ void copy16_g2l(const __bf16* g, __bf16* l) {
#ifdef HAVE_ASYNC_LDS
  __builtin_amdgcn_global_load_async_to_lds_b128((gptr_v4i)g, (lptr_v4i)l, 0, 0);
#else
  *(v8bf*)l = *(const v8bf*)g;
#endif
}

__device__ __forceinline__ void wait_async_lds() {
#ifdef HAVE_ASYNC_LDS
#if __has_builtin(__builtin_amdgcn_s_wait_asynccnt)
  __builtin_amdgcn_s_wait_asynccnt(0);
#else
  asm volatile("s_wait_asynccnt 0" ::: "memory");
#endif
#endif
}

// ---------------------------------------------------------------------------
// WMMA helpers (gfx1250: D = A(16x32 bf16) * B(32x16 bf16) + C(16x16 f32))
// ---------------------------------------------------------------------------
__device__ __forceinline__ v8f wmma_bf16(v16bf a, v16bf b, v8f c) {
  return __builtin_amdgcn_wmma_f32_16x16x32_bf16(
      /*neg_a=*/false, a, /*neg_b=*/false, b,
      /*c_mod=*/(short)0, c, /*reuse_a=*/false, /*reuse_b=*/false);
}

// A fragment: 16x32 (MxK), row-major source with leading dim ld.
// Lanes 0-15 row M=lane: elems 0-7 -> K=kh*8+0..7, elems 8-15 -> K=kh*8+16..23
__device__ __forceinline__ v16bf load_frag_a(const __bf16* base, int ld) {
  const int lane = threadIdx.x & 31;
  const int r  = lane & 15;
  const int kh = lane >> 4;
  const __bf16* p = base + (size_t)r * ld + kh * 8;
  v16bf f;
#pragma unroll
  for (int i = 0; i < 8; ++i) f[i] = p[i];
#pragma unroll
  for (int i = 0; i < 8; ++i) f[8 + i] = p[16 + i];
  return f;
}

// B fragment: 32x16 (KxN) from TRANSPOSED image (N x K row-major, leading dim ld).
// Lanes 0-15 col N=lane: elems 0-7 -> K=kh*8+0..7, elems 8-15 -> K=kh*8+16..23
__device__ __forceinline__ v16bf load_frag_b(const __bf16* baseT, int ld) {
  const int lane = threadIdx.x & 31;
  const int n  = lane & 15;
  const int kh = lane >> 4;
  const __bf16* p = baseT + (size_t)n * ld + kh * 8;
  v16bf f;
#pragma unroll
  for (int i = 0; i < 8; ++i) f[i] = p[i];
#pragma unroll
  for (int i = 0; i < 8; ++i) f[8 + i] = p[16 + i];
  return f;
}

// ---------------------------------------------------------------------------
// fp32 -> bf16 conversion (straight copy)
// ---------------------------------------------------------------------------
__global__ void f32_to_bf16_kernel(const float* __restrict__ src,
                                   __bf16* __restrict__ dst, int n) {
  for (int i = blockIdx.x * blockDim.x + threadIdx.x; i < n;
       i += gridDim.x * blockDim.x)
    dst[i] = (__bf16)src[i];
}

// ---------------------------------------------------------------------------
// fp32 -> bf16 transpose: dst[c][r] = src[r][c], tiled through LDS.
// src: rows x cols (f32). dst: cols x rows (bf16). Block 256 = 32x8.
// ---------------------------------------------------------------------------
__global__ __launch_bounds__(256)
void transpose_f32_bf16_kernel(const float* __restrict__ src,
                               __bf16* __restrict__ dst, int rows, int cols) {
  __shared__ float tile[32][33];
  const int bx = blockIdx.x * 32;  // col tile
  const int by = blockIdx.y * 32;  // row tile
  const int tx = threadIdx.x & 31;
  const int ty = threadIdx.x >> 5;
#pragma unroll
  for (int i = ty; i < 32; i += 8)
    tile[i][tx] = src[(size_t)(by + i) * cols + bx + tx];
  __syncthreads();
#pragma unroll
  for (int i = ty; i < 32; i += 8)
    dst[(size_t)(bx + i) * rows + by + tx] = (__bf16)tile[tx][i];
}

// ---------------------------------------------------------------------------
// Tiled bf16 GEMM, B pre-transposed:  out[M x Nc] = A[M x K] @ Bt[Nc x K]^T + bias
// Block: 256 threads (8 waves). Tile 128x128, BK=32, double-buffered async LDS.
// OUT_MODE: 0 = bf16 row-major, 1 = f32 row-major, 2 = bf16 transposed [Nc x M]
// ---------------------------------------------------------------------------
template <int OUT_MODE>
__global__ __launch_bounds__(256)
void gemm_bf16_kernel(const __bf16* __restrict__ A, const __bf16* __restrict__ Bt,
                      const float* __restrict__ bias, void* __restrict__ outp,
                      int M, int Nc, int K) {
  __shared__ __bf16 Alds[2][128 * 32];
  __shared__ __bf16 Blds[2][128 * 32];

  const int tid  = threadIdx.x;
  const int wave = tid >> 5;
  const int lane = tid & 31;
  const int ln   = lane & 15;
  const int half = lane >> 4;
  const int m0 = blockIdx.y * 128;
  const int n0 = blockIdx.x * 128;

  auto stage = [&](int buf, int k0) {
    // Both tiles are plain 128x32 row-major sub-tiles: 512 16-byte chunks each.
    for (int c = tid; c < 512; c += 256) {
      const int r  = c >> 2;
      const int k8 = (c & 3) * 8;
      copy16_g2l(&A[(size_t)(m0 + r) * K + k0 + k8], &Alds[buf][r * 32 + k8]);
    }
    for (int c = tid; c < 512; c += 256) {
      const int r  = c >> 2;
      const int k8 = (c & 3) * 8;
      copy16_g2l(&Bt[(size_t)(n0 + r) * K + k0 + k8], &Blds[buf][r * 32 + k8]);
    }
  };

  v8f acc[8] = {};

  stage(0, 0);
  wait_async_lds();
  __syncthreads();

  int cur = 0;
  for (int k0 = 0; k0 < K; k0 += 32) {
    if (k0 + 32 < K) stage(cur ^ 1, k0 + 32);  // async prefetch of next tile

    const v16bf af = load_frag_a(&Alds[cur][(wave * 16) * 32], 32);
#pragma unroll
    for (int n = 0; n < 8; ++n) {
      const v16bf bf = load_frag_b(&Blds[cur][(n * 16) * 32], 32);
      acc[n] = wmma_bf16(af, bf, acc[n]);
    }

    wait_async_lds();
    __syncthreads();
    cur ^= 1;
  }

  // Epilogue: C layout -> row = m0 + 16*wave + e + 8*half, col = n0 + 16*n + ln
#pragma unroll
  for (int n = 0; n < 8; ++n) {
    const int col = n0 + n * 16 + ln;
    const float bv = bias ? bias[col] : 0.0f;
#pragma unroll
    for (int e = 0; e < 8; ++e) {
      const int row = m0 + wave * 16 + 8 * half + e;
      const float v = acc[n][e] + bv;
      if (OUT_MODE == 1)
        ((float*)outp)[(size_t)row * Nc + col] = v;
      else if (OUT_MODE == 2)
        ((__bf16*)outp)[(size_t)col * M + row] = (__bf16)v;  // transposed store
      else
        ((__bf16*)outp)[(size_t)row * Nc + col] = (__bf16)v;
    }
  }
}

// ---------------------------------------------------------------------------
// Flash attention: grid (row-blocks, heads). 4 waves / block, 16 rows / wave.
// K tile [kv][d] and Vt tile [d][kv] are straight row-major copies (V comes
// pre-transposed from the projection GEMM), double-buffered via async LDS.
// ---------------------------------------------------------------------------
__global__ __launch_bounds__(128)
void flash_attn_kernel(const __bf16* __restrict__ Q, const __bf16* __restrict__ Km,
                       const __bf16* __restrict__ Vt, const float* __restrict__ mask,
                       __bf16* __restrict__ Out) {
  __shared__ __bf16 Klds[2][64 * 128];   // [kv][d]
  __shared__ __bf16 Vlds[2][128 * 64];   // [d][kv]
  __shared__ __bf16 Plds[4][16 * 64];    // per-wave P, row-major [m][kv]

  const int h    = blockIdx.y;
  const int tid  = threadIdx.x;
  const int wave = tid >> 5;
  const int lane = tid & 31;
  const int ln   = lane & 15;
  const int half = lane >> 4;
  const int row0 = blockIdx.x * 64 + wave * 16;
  const int hoff = h * D_HEAD;

  auto stage = [&](int buf, int kv0) {
    // K tile: 64 rows x 128 (16 chunks/row)
    for (int c = tid; c < 1024; c += 128) {
      const int r  = c >> 4;
      const int c8 = (c & 15) * 8;
      copy16_g2l(&Km[(size_t)(kv0 + r) * D_MODEL + hoff + c8],
                 &Klds[buf][r * 128 + c8]);
    }
    // Vt tile: 128 rows (d) x 64 (kv) (8 chunks/row); Vt is [D_MODEL][N_TOK]
    for (int c = tid; c < 1024; c += 128) {
      const int r  = c >> 3;
      const int c8 = (c & 7) * 8;
      copy16_g2l(&Vt[(size_t)(hoff + r) * N_TOK + kv0 + c8],
                 &Vlds[buf][r * 64 + c8]);
    }
  };

  // Q fragments for this wave's 16 rows (K-dim = 128 -> 4 frags)
  v16bf qf[4];
  {
    const __bf16* qbase = Q + (size_t)row0 * D_MODEL + hoff;
#pragma unroll
    for (int kk = 0; kk < 4; ++kk) qf[kk] = load_frag_a(qbase + kk * 32, D_MODEL);
  }

  v8f o[8] = {};
  float mrow[8], lrow[8];
#pragma unroll
  for (int e = 0; e < 8; ++e) { mrow[e] = -3.0e38f; lrow[e] = 0.0f; }

  const float inv_scale = 0.03125f;  // 1/sqrt(1024)

  stage(0, 0);
  wait_async_lds();
  __syncthreads();

  int cur = 0;
  for (int kv0 = 0; kv0 < N_TOK; kv0 += 64) {
    if (kv0 + 64 < N_TOK) stage(cur ^ 1, kv0 + 64);  // async prefetch

    // S = Q @ K^T  (16 x 64): 4 n-subtiles x 4 k-steps
    v8f sacc[4];
#pragma unroll
    for (int s = 0; s < 4; ++s) {
      v8f a = {};
#pragma unroll
      for (int kk = 0; kk < 4; ++kk) {
        const v16bf bf = load_frag_b(&Klds[cur][(s * 16) * 128 + kk * 32], 128);
        a = wmma_bf16(qf[kk], bf, a);
      }
      sacc[s] = a;
    }

    // scale + additive mask
#pragma unroll
    for (int s = 0; s < 4; ++s)
#pragma unroll
      for (int e = 0; e < 8; ++e) {
        const int r = row0 + 8 * half + e;
        const int cidx = kv0 + s * 16 + ln;
        sacc[s][e] = sacc[s][e] * inv_scale + mask[(size_t)r * N_TOK + cidx];
      }

    // row max (over s, then over the 16 lanes of this half)
    float tmax[8];
#pragma unroll
    for (int e = 0; e < 8; ++e) {
      float m = sacc[0][e];
#pragma unroll
      for (int s = 1; s < 4; ++s) m = fmaxf(m, sacc[s][e]);
#pragma unroll
      for (int d = 8; d >= 1; d >>= 1) m = fmaxf(m, __shfl_xor(m, d, 16));
      tmax[e] = m;
    }

    float alpha[8];
#pragma unroll
    for (int e = 0; e < 8; ++e) {
      const float mn = fmaxf(mrow[e], tmax[e]);
      alpha[e] = __expf(mrow[e] - mn);
      mrow[e] = mn;
    }

    // P = exp(S - m), row sums
    float rsum[8] = {};
#pragma unroll
    for (int s = 0; s < 4; ++s)
#pragma unroll
      for (int e = 0; e < 8; ++e) {
        const float p = __expf(sacc[s][e] - mrow[e]);
        sacc[s][e] = p;
        rsum[e] += p;
      }
#pragma unroll
    for (int e = 0; e < 8; ++e) {
      float r = rsum[e];
#pragma unroll
      for (int d = 8; d >= 1; d >>= 1) r += __shfl_xor(r, d, 16);
      lrow[e] = lrow[e] * alpha[e] + r;
    }

    // rescale O
#pragma unroll
    for (int n = 0; n < 8; ++n)
#pragma unroll
      for (int e = 0; e < 8; ++e) o[n][e] *= alpha[e];

    // P (C layout) -> wave-private LDS, row-major [16][64]
    __bf16* pl = &Plds[wave][0];
#pragma unroll
    for (int s = 0; s < 4; ++s)
#pragma unroll
      for (int e = 0; e < 8; ++e)
        pl[(8 * half + e) * 64 + s * 16 + ln] = (__bf16)sacc[s][e];
    // same-wave DS ops are in order; compiler inserts s_wait_dscnt for the loads

    v16bf pa[2];
#pragma unroll
    for (int kk = 0; kk < 2; ++kk) pa[kk] = load_frag_a(pl + kk * 32, 64);

    // O += P @ V : 8 n-subtiles (d) x 2 k-steps (kv)
#pragma unroll
    for (int n = 0; n < 8; ++n)
#pragma unroll
      for (int kk = 0; kk < 2; ++kk) {
        const v16bf bv = load_frag_b(&Vlds[cur][(n * 16) * 64 + kk * 32], 64);
        o[n] = wmma_bf16(pa[kk], bv, o[n]);
      }

    wait_async_lds();
    __syncthreads();
    cur ^= 1;
  }

  // Epilogue: Out[row, h*128 + d] = O / l  (head-major concat layout)
#pragma unroll
  for (int n = 0; n < 8; ++n)
#pragma unroll
    for (int e = 0; e < 8; ++e) {
      const int row = row0 + 8 * half + e;
      const int col = hoff + n * 16 + ln;
      Out[(size_t)row * D_MODEL + col] = (__bf16)(o[n][e] / lrow[e]);
    }
}

// ---------------------------------------------------------------------------
// Host-side launch
// ---------------------------------------------------------------------------
extern "C" void kernel_launch(void* const* d_in, const int* in_sizes, int n_in,
                              void* d_out, int out_size, void* d_ws, size_t ws_size,
                              hipStream_t stream) {
  (void)in_sizes; (void)n_in; (void)out_size; (void)ws_size;
  const float* x    = (const float*)d_in[0];
  const float* mask = (const float*)d_in[1];
  const float* W_Q  = (const float*)d_in[2];
  const float* W_K  = (const float*)d_in[3];
  const float* W_V  = (const float*)d_in[4];
  const float* W_O  = (const float*)d_in[5];
  float* out = (float*)d_out;

  char* ws = (char*)d_ws;
  const size_t SZ_X = (size_t)N_TOK * D_MODEL * sizeof(__bf16);    // 8 MiB
  const size_t SZ_W = (size_t)D_MODEL * D_MODEL * sizeof(__bf16);  // 2 MiB
  __bf16* Xb   = (__bf16*)(ws);
  __bf16* WqT  = (__bf16*)(ws + SZ_X);
  __bf16* WkT  = (__bf16*)(ws + SZ_X + SZ_W);
  __bf16* WvT  = (__bf16*)(ws + SZ_X + 2 * SZ_W);
  __bf16* WoT  = (__bf16*)(ws + SZ_X + 3 * SZ_W);
  __bf16* Qb   = (__bf16*)(ws + SZ_X + 4 * SZ_W);
  __bf16* Kb   = (__bf16*)(ws + 2 * SZ_X + 4 * SZ_W);
  __bf16* VbT  = (__bf16*)(ws + 3 * SZ_X + 4 * SZ_W);  // [D_MODEL][N_TOK]
  __bf16* Attn = (__bf16*)(ws + 4 * SZ_X + 4 * SZ_W);

  const int NX = N_TOK * D_MODEL;  // 4M
  f32_to_bf16_kernel<<<(NX + 255) / 256, 256, 0, stream>>>(x, Xb, NX);

  // Transpose-convert weights (first 1024 rows; row 1024 = fp32 bias)
  dim3 tgrid(D_MODEL / 32, D_MODEL / 32);  // (32, 32)
  transpose_f32_bf16_kernel<<<tgrid, 256, 0, stream>>>(W_Q, WqT, D_MODEL, D_MODEL);
  transpose_f32_bf16_kernel<<<tgrid, 256, 0, stream>>>(W_K, WkT, D_MODEL, D_MODEL);
  transpose_f32_bf16_kernel<<<tgrid, 256, 0, stream>>>(W_V, WvT, D_MODEL, D_MODEL);
  transpose_f32_bf16_kernel<<<tgrid, 256, 0, stream>>>(W_O, WoT, D_MODEL, D_MODEL);

  // Projections: [4096x1024] @ W + bias. V is written transposed for the
  // attention kernel's async-friendly [d][token] tile layout.
  dim3 ggrid(D_MODEL / 128, N_TOK / 128);  // (8, 32)
  gemm_bf16_kernel<0><<<ggrid, 256, 0, stream>>>(
      Xb, WqT, W_Q + (size_t)D_MODEL * D_MODEL, Qb, N_TOK, D_MODEL, D_MODEL);
  gemm_bf16_kernel<0><<<ggrid, 256, 0, stream>>>(
      Xb, WkT, W_K + (size_t)D_MODEL * D_MODEL, Kb, N_TOK, D_MODEL, D_MODEL);
  gemm_bf16_kernel<2><<<ggrid, 256, 0, stream>>>(
      Xb, WvT, W_V + (size_t)D_MODEL * D_MODEL, VbT, N_TOK, D_MODEL, D_MODEL);

  // Attention (flash, online softmax)
  dim3 fgrid(N_TOK / 64, N_HEADS);  // (64, 8)
  flash_attn_kernel<<<fgrid, 128, 0, stream>>>(Qb, Kb, VbT, mask, Attn);

  // Output projection -> fp32 d_out
  gemm_bf16_kernel<1><<<ggrid, 256, 0, stream>>>(
      Attn, WoT, W_O + (size_t)D_MODEL * D_MODEL, out, N_TOK, D_MODEL, D_MODEL);
}
